// SelfAttention_69965017252228
// MI455X (gfx1250) — compile-verified, matrix operand order checked by reading, not run on hardware
//
#include <hip/hip_runtime.h>

// Problem constants (B=4, L=1024, D=256, H=8)
#define BB   4
#define LL   1024
#define DD   256
#define HH   8
#define HD   2048       // H*D
#define MTOT 4096       // B*L

typedef __attribute__((ext_vector_type(16))) __bf16 v16bf;
typedef __attribute__((ext_vector_type(8)))  __bf16 v8bf;
typedef __attribute__((ext_vector_type(8)))  float  v8f;

static __device__ __forceinline__ v8f wmma_bf16(v16bf a, v16bf b, v8f c) {
  // D = A(16x32 bf16) * B(32x16 bf16) + C(16x16 f32)
  return __builtin_amdgcn_wmma_f32_16x16x32_bf16(false, a, false, b, (short)0, c,
                                                 false, false);
}

// A/B fragment, reduction dim contiguous in memory, f32 source -> cvt bf16.
// Lane layout (wave32): half = lane>>4; elems 0..7 at k = half*8 + j,
// elems 8..15 at k = 16 + half*8 + j.
static __device__ __forceinline__ v16bf frag_from_f32(const float* p, int half) {
  v8f lo = *(const v8f*)(p + half * 8);
  v8f hi = *(const v8f*)(p + 16 + half * 8);
  v16bf r;
#pragma unroll
  for (int j = 0; j < 8; ++j) { r[j] = (__bf16)lo[j]; r[j + 8] = (__bf16)hi[j]; }
  return r;
}

static __device__ __forceinline__ v16bf frag_from_bf16(const __bf16* p, int half) {
  v8bf lo = *(const v8bf*)(p + half * 8);
  v8bf hi = *(const v8bf*)(p + 16 + half * 8);
  return __builtin_shufflevector(lo, hi, 0, 1, 2, 3, 4, 5, 6, 7,
                                 8, 9, 10, 11, 12, 13, 14, 15);
}

static __device__ __forceinline__ float wave_max(float v) {
#pragma unroll
  for (int o = 16; o > 0; o >>= 1) v = fmaxf(v, __shfl_xor(v, o, 32));
  return v;
}
static __device__ __forceinline__ float wave_sum(float v) {
#pragma unroll
  for (int o = 16; o > 0; o >>= 1) v += __shfl_xor(v, o, 32);
  return v;
}

// ---------------------------------------------------------------------------
// Kernel 1: projections  Y = X @ W^T  for (q,wq)->qh, (k,wk)->kh, (v,wv)->vh^T
// X:[4096,256] f32, W:[2048,256] f32. qh/kh stored bf16 [B,L,H,D]=[4096,2048];
// vh stored TRANSPOSED bf16 [B,H,D,L] so attn@V B-fragments are contiguous.
// grid (256, 4, 3), block 256 (8 waves); each wave: 16(M) x 64(N) tile.
// ---------------------------------------------------------------------------
__global__ __launch_bounds__(256) void k_proj(
    const float* __restrict__ q, const float* __restrict__ k,
    const float* __restrict__ v, const float* __restrict__ wq,
    const float* __restrict__ wk, const float* __restrict__ wv,
    __bf16* __restrict__ qh, __bf16* __restrict__ kh, __bf16* __restrict__ vt) {
  const int lane = threadIdx.x & 31;
  const int wave = threadIdx.x >> 5;
  const int half = lane >> 4;
  const int lrow = lane & 15;
  const int m0 = blockIdx.x * 16;
  const int n0 = blockIdx.y * 512 + wave * 64;
  const int z  = blockIdx.z;

  const float* X = (z == 0) ? q : (z == 1) ? k : v;
  const float* W = (z == 0) ? wq : (z == 1) ? wk : wv;

  const v8f vzero = {};
  v8f acc[4];
#pragma unroll
  for (int t = 0; t < 4; ++t) acc[t] = vzero;

  const float* arow = X + (size_t)(m0 + lrow) * DD;
#pragma unroll
  for (int kb = 0; kb < DD; kb += 32) {
    v16bf af = frag_from_f32(arow + kb, half);
#pragma unroll
    for (int t = 0; t < 4; ++t) {
      const float* brow = W + (size_t)(n0 + 16 * t + lrow) * DD;
      v16bf bf = frag_from_f32(brow + kb, half);
      acc[t] = wmma_bf16(af, bf, acc[t]);
    }
  }

  if (z < 2) {
    __bf16* Y = (z == 0) ? qh : kh;
#pragma unroll
    for (int t = 0; t < 4; ++t) {
      int n = n0 + 16 * t + lrow;
#pragma unroll
      for (int r = 0; r < 8; ++r) {
        int m = m0 + 8 * half + r;
        Y[(size_t)m * HD + n] = (__bf16)acc[t][r];
      }
    }
  } else {
    // transposed store: vt[((b*H + h)*D + d)*L + kpos], 8 consecutive kpos/lane
    int b  = m0 >> 10;
    int k0 = (m0 & 1023) + 8 * half;
#pragma unroll
    for (int t = 0; t < 4; ++t) {
      int n = n0 + 16 * t + lrow;
      int h = n >> 8, d = n & 255;
      v8bf pk;
#pragma unroll
      for (int r = 0; r < 8; ++r) pk[r] = (__bf16)acc[t][r];
      *(v8bf*)(vt + ((size_t)((b * HH + h) * DD + d)) * LL + k0) = pk;
    }
  }
}

// ---------------------------------------------------------------------------
// Kernel 2: scores = (qh . kh)/16 for one (b,h,q-tile of 16), then the exact
// masked softmax of the reference; writes f32 attn to d_out in [H*B,L,L].
// grid (64, 8, 4), block 256.
//
// The 16x256 bf16 qh tile is shared by ALL 8 waves: stage it into LDS once
// via global_load_async_to_lds_b128 (ASYNCcnt DMA path), then build A
// fragments from LDS (ds_load_b128). The 64KB score buffer is only live after
// the WMMA loop, so the 8KB staged tile reuses its front bytes.
// ---------------------------------------------------------------------------
__global__ __launch_bounds__(256) void k_attn(
    const __bf16* __restrict__ qh, const __bf16* __restrict__ kh,
    const int* __restrict__ mask, float* __restrict__ attn) {
  __shared__ float sm[16 * LL];   // 64 KB scores; front 8 KB doubles as qh tile
  __bf16* smq = (__bf16*)sm;      // [16][256] bf16 staged A tile

  const int lane = threadIdx.x & 31;
  const int wave = threadIdx.x >> 5;
  const int half = lane >> 4;
  const int lrow = lane & 15;
  const int q0 = blockIdx.x * 16;
  const int h  = blockIdx.y;
  const int b  = blockIdx.z;
  const int n0 = wave * 128;

  // ---- phase 0: async-stage qh tile (16 rows x 256 bf16 = 512 x 16B chunks)
  {
    int c = threadIdx.x;
#pragma unroll
    for (int it = 0; it < 2; ++it, c += 256) {
      int row = c >> 5;            // 0..15
      int off = (c & 31) * 8;      // bf16 element offset within row
      unsigned ldsa = (unsigned)(size_t)(smq + row * DD + off);
      const __bf16* gsrc = qh + (size_t)(b * LL + q0 + row) * HD + h * DD + off;
      asm volatile("global_load_async_to_lds_b128 %0, %1, off"
                   :
                   : "v"(ldsa), "v"(gsrc)
                   : "memory");
    }
    asm volatile("s_wait_asynccnt 0" ::: "memory");
  }
  __syncthreads();

  // ---- phase 1: QK^T with A from LDS, B from global bf16
  const v8f vzero = {};
  v8f acc[8];
#pragma unroll
  for (int t = 0; t < 8; ++t) acc[t] = vzero;

  const __bf16* aro = smq + lrow * DD;   // LDS: row lrow, contiguous d
#pragma unroll
  for (int kb = 0; kb < DD; kb += 32) {
    v16bf af = frag_from_bf16(aro + kb, half);
#pragma unroll
    for (int t = 0; t < 8; ++t) {
      const __bf16* bro =
          kh + (size_t)(b * LL + n0 + 16 * t + lrow) * HD + h * DD;
      v16bf bf = frag_from_bf16(bro + kb, half);
      acc[t] = wmma_bf16(af, bf, acc[t]);
    }
  }
  __syncthreads();  // done reading staged tile; sm becomes the score buffer

  const float inv_temp = 1.0f / 16.0f;  // 1/sqrt(D)
#pragma unroll
  for (int t = 0; t < 8; ++t) {
    int c = n0 + 16 * t + lrow;
#pragma unroll
    for (int r = 0; r < 8; ++r)
      sm[(8 * half + r) * LL + c] = acc[t][r] * inv_temp;
  }
  __syncthreads();

  // ---- phase 2: masked softmax, x=clip(s,-15,15)*m; e=exp(x-max)*m; e/(sum+eps)
  for (int rr = 2 * wave; rr < 2 * wave + 2; ++rr) {
    const int qrow = q0 + rr;
    const int* mrow = mask + ((size_t)b * LL + qrow) * LL;
    float mx = -1e30f;
    for (int c = lane; c < LL; c += 32) {
      float mf = (float)mrow[c];
      float x = fminf(fmaxf(sm[rr * LL + c], -15.0f), 15.0f) * mf;
      sm[rr * LL + c] = x;
      mx = fmaxf(mx, x);
    }
    mx = wave_max(mx);
    float s = 0.0f;
    for (int c = lane; c < LL; c += 32) {
      float mf = (float)mrow[c];
      float e = __expf(sm[rr * LL + c] - mx) * mf;
      sm[rr * LL + c] = e;
      s += e;
    }
    s = wave_sum(s);
    float inv = 1.0f / (s + 1e-6f);
    float* orow = attn + ((size_t)(h * BB + b) * LL + qrow) * LL;
    for (int c = lane; c < LL; c += 32) orow[c] = sm[rr * LL + c] * inv;
  }
}

// ---------------------------------------------------------------------------
// Kernel 3: out_heads[b,q,h*D+d] = sum_k attn[h,b,q,k] * v^T[b,h,d,k]
// A = f32 attn rows (contiguous k), B = bf16 vt rows (contiguous k).
// grid (64, 8, 4), block 256; each wave does 16(q) x 32(d).
// ---------------------------------------------------------------------------
__global__ __launch_bounds__(256) void k_av(
    const float* __restrict__ attn, const __bf16* __restrict__ vt,
    float* __restrict__ oh) {
  const int lane = threadIdx.x & 31;
  const int wave = threadIdx.x >> 5;
  const int half = lane >> 4;
  const int lrow = lane & 15;
  const int q0 = blockIdx.x * 16;
  const int h  = blockIdx.y;
  const int b  = blockIdx.z;
  const int d0 = wave * 32;

  const v8f vzero = {};
  v8f acc[2] = {vzero, vzero};
  const float* aro = attn + ((size_t)(h * BB + b) * LL + q0 + lrow) * LL;
  for (int kb = 0; kb < LL; kb += 32) {
    v16bf af = frag_from_f32(aro + kb, half);
#pragma unroll
    for (int t = 0; t < 2; ++t) {
      const __bf16* bro =
          vt + ((size_t)((b * HH + h) * DD) + d0 + 16 * t + lrow) * LL;
      v16bf bf = frag_from_bf16(bro + kb, half);
      acc[t] = wmma_bf16(af, bf, acc[t]);
    }
  }
#pragma unroll
  for (int t = 0; t < 2; ++t) {
    int d = d0 + 16 * t + lrow;
#pragma unroll
    for (int r = 0; r < 8; ++r) {
      int m = b * LL + q0 + 8 * half + r;
      oh[(size_t)m * HD + h * DD + d] = acc[t][r];
    }
  }
}

// ---------------------------------------------------------------------------
// Kernel 4: out = LN(out_heads @ fc_w^T + fc_b + residual) * g + beta
// grid (256), block 256; block owns 16 rows x 256 cols, waves do 16x32 GEMM
// tiles into LDS, then 2-rows-per-wave LayerNorm over D=256.
// ---------------------------------------------------------------------------
__global__ __launch_bounds__(256) void k_fc_ln(
    const float* __restrict__ oh, const float* __restrict__ fcw,
    const float* __restrict__ fcb, const float* __restrict__ resid,
    const float* __restrict__ g, const float* __restrict__ beta,
    float* __restrict__ out) {
  __shared__ float sm[16 * DD];  // 16 KB
  const int lane = threadIdx.x & 31;
  const int wave = threadIdx.x >> 5;
  const int half = lane >> 4;
  const int lrow = lane & 15;
  const int m0 = blockIdx.x * 16;
  const int nb = wave * 32;

  const v8f vzero = {};
  v8f acc[2] = {vzero, vzero};
  const float* aro = oh + (size_t)(m0 + lrow) * HD;
  for (int kb = 0; kb < HD; kb += 32) {
    v16bf af = frag_from_f32(aro + kb, half);
#pragma unroll
    for (int t = 0; t < 2; ++t) {
      const float* bro = fcw + (size_t)(nb + 16 * t + lrow) * HD;
      __builtin_prefetch(bro + kb + 64, 0, 1);
      v16bf bf = frag_from_f32(bro + kb, half);
      acc[t] = wmma_bf16(af, bf, acc[t]);
    }
  }
#pragma unroll
  for (int t = 0; t < 2; ++t) {
    int n = nb + 16 * t + lrow;
    float bias = fcb[n];
#pragma unroll
    for (int r = 0; r < 8; ++r) {
      int mr = 8 * half + r;
      sm[mr * DD + n] = acc[t][r] + bias + resid[(size_t)(m0 + mr) * DD + n];
    }
  }
  __syncthreads();

  for (int rr = 2 * wave; rr < 2 * wave + 2; ++rr) {
    float s = 0.0f, ss = 0.0f;
    for (int c = lane; c < DD; c += 32) {
      float v = sm[rr * DD + c];
      s += v;
      ss += v * v;
    }
    s = wave_sum(s);
    ss = wave_sum(ss);
    float mean = s * (1.0f / DD);
    float var = ss * (1.0f / DD) - mean * mean;
    float rstd = rsqrtf(var + 1e-5f);
    for (int c = lane; c < DD; c += 32) {
      float v = sm[rr * DD + c];
      out[(size_t)(m0 + rr) * DD + c] = (v - mean) * rstd * g[c] + beta[c];
    }
  }
}

// ---------------------------------------------------------------------------
extern "C" void kernel_launch(void* const* d_in, const int* in_sizes, int n_in,
                              void* d_out, int out_size, void* d_ws,
                              size_t ws_size, hipStream_t stream) {
  const float* q    = (const float*)d_in[0];
  const int*   mask = (const int*)d_in[1];
  const float* k    = (const float*)d_in[2];
  const float* v    = (const float*)d_in[3];
  const float* wq   = (const float*)d_in[4];
  const float* wk   = (const float*)d_in[5];
  const float* wv   = (const float*)d_in[6];
  const float* fcw  = (const float*)d_in[7];
  const float* fcb  = (const float*)d_in[8];
  const float* lng  = (const float*)d_in[9];
  const float* lnb  = (const float*)d_in[10];

  float* out  = (float*)d_out;                  // [B,L,D]
  float* attn = out + (size_t)MTOT * DD;        // [H*B,L,L]

  // workspace layout (80 MB total):
  __bf16* qh = (__bf16*)d_ws;                   // 16 MB [B,L,H,D] bf16
  __bf16* kh = qh + (size_t)MTOT * HD;          // 16 MB [B,L,H,D] bf16
  __bf16* vt = kh + (size_t)MTOT * HD;          // 16 MB [B,H,D,L] bf16
  float*  oh = (float*)(vt + (size_t)MTOT * HD);// 32 MB [B,L,H*D] f32

  k_proj<<<dim3(MTOT / 16, 4, 3), 256, 0, stream>>>(q, k, v, wq, wk, wv, qh, kh,
                                                    vt);
  k_attn<<<dim3(LL / 16, HH, BB), 256, 0, stream>>>(qh, kh, mask, attn);
  k_av<<<dim3(LL / 16, HH, BB), 256, 0, stream>>>(attn, vt, oh);
  k_fc_ln<<<dim3(MTOT / 16, 1, 1), 256, 0, stream>>>(oh, fcw, fcb, q, lng, lnb,
                                                     out);
}
